// transformer_model_83159156785839
// MI455X (gfx1250) — matrix operand order repaired and checked by reading, not compiled, not run on hardware
//
#include <hip/hip_runtime.h>
#include <stdint.h>
#include <math.h>

// ---------------- problem constants (from reference) ----------------
#define BATCH 16
#define SEQ   512
#define DIM   9984      // 13*768
#define HID1  1000
#define HID2  40
#define NTOK  (BATCH*SEQ)   // 8192 rows
#define W1PAD 1024          // padded N stride of pre-split W1 (zero-filled tail)

// ---------------- GEMM1 tiling ----------------
#define TM 128
#define TN 128
#define TK 32
#define NKT (DIM/TK)  // 312
#define SA 40    // ushort stride of A LDS rows (80B, 16B aligned, bank stride 20)
#define SB 136   // ushort stride of B LDS rows (272B, 16B aligned)

typedef __attribute__((ext_vector_type(16))) __bf16 v16bf;
typedef __attribute__((ext_vector_type(8)))  float  v8f;

struct alignas(16) F4 { float x, y, z, w; };

__device__ __forceinline__ unsigned short bf16_rne(float x) {
    unsigned u = __float_as_uint(x);
    unsigned r = u + 0x7FFFu + ((u >> 16) & 1u);   // round-to-nearest-even
    return (unsigned short)(r >> 16);
}

__device__ __forceinline__ void split_bf16(float x, unsigned short& h, unsigned short& l) {
    h = bf16_rne(x);
    float fh = __uint_as_float(((unsigned)h) << 16);
    l = bf16_rne(x - fh);
}

// fast truncation split of two floats -> packed bf16 hi pair + lo pair.
// hi = trunc16(x); r = x - hi is EXACT in fp32, lo = trunc16(r);
// dropped term <= 2^-16 |x|  (same bound as RNE split), 3 VALU/float.
__device__ __forceinline__ void split2_pack(float x, float y,
                                            unsigned& hpack, unsigned& lpack) {
    unsigned ux = __float_as_uint(x), uy = __float_as_uint(y);
    float fx = __uint_as_float(ux & 0xFFFF0000u);
    float fy = __uint_as_float(uy & 0xFFFF0000u);
    float rx = x - fx, ry = y - fy;
    // select bytes {y[3],y[2],x[3],x[2]} -> lo half = hi16(x), hi half = hi16(y)
    hpack = __builtin_amdgcn_perm(uy, ux, 0x07060302u);
    lpack = __builtin_amdgcn_perm(__float_as_uint(ry), __float_as_uint(rx), 0x07060302u);
}

// load 16 bf16 (two 16B chunks) from LDS into a v16bf fragment
__device__ __forceinline__ v16bf frag_ld(const unsigned short* p, int off2_ushorts) {
    union { unsigned u[8]; v16bf v; } r;
    const unsigned* q0 = (const unsigned*)(p);
    const unsigned* q1 = (const unsigned*)(p + off2_ushorts);
    r.u[0] = q0[0]; r.u[1] = q0[1]; r.u[2] = q0[2]; r.u[3] = q0[3];
    r.u[4] = q1[0]; r.u[5] = q1[1]; r.u[6] = q1[2]; r.u[7] = q1[3];
    return r.v;
}

// CDNA5 async global->LDS copy, 16B per lane, GVS addressing (ASYNCcnt tracked)
__device__ __forceinline__ void async_copy_b128(unsigned lds_byte_addr,
                                                const unsigned short* sbase,
                                                unsigned gbyte_off) {
    asm volatile("global_load_async_to_lds_b128 %0, %1, %2"
                 :: "v"(lds_byte_addr), "v"(gbyte_off), "s"(sbase)
                 : "memory");
}

__device__ __forceinline__ void wait_asynccnt0() {
    asm volatile("s_wait_asynccnt 0" ::: "memory");
}

// ---------------------------------------------------------------------------
// Kernel 0: pre-split W1 fp32 -> bf16 hi/lo with padded (zeroed) N stride 1024
// ---------------------------------------------------------------------------
__global__ __launch_bounds__(256) void split_w1_kernel(
    const float* __restrict__ W1, unsigned short* __restrict__ Wh,
    unsigned short* __restrict__ Wl)
{
    int id = blockIdx.x * 256 + threadIdx.x;      // over 9984*1024
    if (id >= DIM * W1PAD) return;
    int k = id >> 10, n = id & (W1PAD - 1);
    float v = (n < HID1) ? W1[(size_t)k * HID1 + n] : 0.0f;
    unsigned short h, l;
    split_bf16(v, h, l);
    Wh[id] = h;
    Wl[id] = l;
}

// ---------------------------------------------------------------------------
// Kernel 1: H = relu(X * W1 + b1), split-bf16 fp32 emulation
//   - B tiles: pure bf16 copies via global_load_async_to_lds_b128
//   - A tiles: global->VGPR prefetch, trunc-split + v_perm pack, LDS store
//   - double-buffered LDS ping-pong, one barrier per k-tile
// ---------------------------------------------------------------------------
__global__ __launch_bounds__(256) void gemm1_relu_kernel(
    const float* __restrict__ X,
    const unsigned short* __restrict__ Wh, const unsigned short* __restrict__ Wl,
    const float* __restrict__ b1, float* __restrict__ Hout)
{
    __shared__ alignas(16) unsigned short Ah[2][TM * SA];
    __shared__ alignas(16) unsigned short Al[2][TM * SA];
    __shared__ alignas(16) unsigned short Bh[2][TK * SB];
    __shared__ alignas(16) unsigned short Bl[2][TK * SB];

    const int t    = threadIdx.x;
    const int lane = t & 31;
    const int wv   = t >> 5;
    const int nb   = blockIdx.x & 7;   // N fastest -> A stripe reused from L2
    const int mb   = blockIdx.x >> 3;
    const int m0   = mb * TM;
    const int n0   = nb * TN;

    // LDS byte addresses (flat low 32 bits == LDS offset) for async writes
    const unsigned bhAddr[2] = { (unsigned)(uintptr_t)&Bh[0][0],
                                 (unsigned)(uintptr_t)&Bh[1][0] };
    const unsigned blAddr[2] = { (unsigned)(uintptr_t)&Bl[0][0],
                                 (unsigned)(uintptr_t)&Bl[1][0] };

    v8f acc[8];
#pragma unroll
    for (int i = 0; i < 8; ++i)
#pragma unroll
        for (int j = 0; j < 8; ++j) acc[i][j] = 0.0f;

    const int arow  = wv * 16 + (lane & 15);
    const int akoff = (lane >> 4) * 8;          // K chunk start per ISA A layout

    // per-thread async B chunk coordinates: 512 16B chunks per tile, 2/thread
    const int bkr0 = t >> 4;            // 0..15  (id = t)
    const int bc0  = t & 15;
    const int bkr1 = (t + 256) >> 4;    // 16..31 (id = t+256)
    const int bc1  = (t + 256) & 15;

    F4 areg[4];

    // -------- prologue: stage k-tile 0 into buffer 0 --------
    {
        const int k0 = 0;
#pragma unroll
        for (int j = 0; j < 2; ++j) {
            int kr = j ? bkr1 : bkr0, c = j ? bc1 : bc0;
            unsigned ldsoff = (unsigned)(kr * (SB * 2) + c * 16);
            unsigned goff   = (unsigned)(((k0 + kr) * W1PAD + n0 + c * 8) * 2);
            async_copy_b128(bhAddr[0] + ldsoff, Wh, goff);
            async_copy_b128(blAddr[0] + ldsoff, Wl, goff);
        }
#pragma unroll
        for (int i = 0; i < 4; ++i) {
            int li = t + i * 256, r = li >> 3, c4 = li & 7;
            areg[i] = *reinterpret_cast<const F4*>(X + (size_t)(m0 + r) * DIM + k0 + c4 * 4);
        }
#pragma unroll
        for (int i = 0; i < 4; ++i) {
            int li = t + i * 256, r = li >> 3, c4 = li & 7;
            int base = r * SA + c4 * 4;                 // even ushort index
            unsigned h0, l0, h1, l1;
            split2_pack(areg[i].x, areg[i].y, h0, l0);
            split2_pack(areg[i].z, areg[i].w, h1, l1);
            unsigned* ah = (unsigned*)&Ah[0][base];
            unsigned* al = (unsigned*)&Al[0][base];
            ah[0] = h0; ah[1] = h1;
            al[0] = l0; al[1] = l1;
        }
        wait_asynccnt0();
        __syncthreads();
    }

    int cur = 0;
    for (int kt = 0; kt < NKT; ++kt) {
        const int nxt = cur ^ 1;
        const bool more = (kt + 1) < NKT;
        const int k0n = (kt + 1) * TK;

        // ---- stage next tile: async B copies + A register prefetch ----
        if (more) {
#pragma unroll
            for (int j = 0; j < 2; ++j) {
                int kr = j ? bkr1 : bkr0, c = j ? bc1 : bc0;
                unsigned ldsoff = (unsigned)(kr * (SB * 2) + c * 16);
                unsigned goff   = (unsigned)(((k0n + kr) * W1PAD + n0 + c * 8) * 2);
                async_copy_b128(bhAddr[nxt] + ldsoff, Wh, goff);
                async_copy_b128(blAddr[nxt] + ldsoff, Wl, goff);
            }
#pragma unroll
            for (int i = 0; i < 4; ++i) {
                int li = t + i * 256, r = li >> 3, c4 = li & 7;
                areg[i] = *reinterpret_cast<const F4*>(X + (size_t)(m0 + r) * DIM + k0n + c4 * 4);
            }
        }

        // ---- compute current tile: 8 subtiles x 3 split-bf16 WMMAs ----
        {
            v16bf a_h = frag_ld(&Ah[cur][arow * SA + akoff], 16);
            v16bf a_l = frag_ld(&Al[cur][arow * SA + akoff], 16);
#pragma unroll
            for (int nt = 0; nt < 8; ++nt) {
                v16bf b_h, b_l;
#pragma unroll
                for (int s = 0; s < 16; ++s) {
                    int kk = akoff + (s & 7) + ((s >> 3) << 4);
                    int bi = kk * SB + nt * 16 + (lane & 15);
                    b_h[s] = __builtin_bit_cast(__bf16, Bh[cur][bi]);
                    b_l[s] = __builtin_bit_cast(__bf16, Bl[cur][bi]);
                }
                acc[nt] = __builtin_amdgcn_wmma_f32_16x16x32_bf16(
                    false, a_h, false, b_h, (short)0, acc[nt], false, false);
                acc[nt] = __builtin_amdgcn_wmma_f32_16x16x32_bf16(
                    false, a_h, false, b_l, (short)0, acc[nt], false, false);
                acc[nt] = __builtin_amdgcn_wmma_f32_16x16x32_bf16(
                    false, a_l, false, b_h, (short)0, acc[nt], false, false);
            }
        }

        // ---- convert + store prefetched A into next buffer ----
        if (more) {
#pragma unroll
            for (int i = 0; i < 4; ++i) {
                int li = t + i * 256, r = li >> 3, c4 = li & 7;
                int base = r * SA + c4 * 4;
                unsigned h0, l0, h1, l1;
                split2_pack(areg[i].x, areg[i].y, h0, l0);
                split2_pack(areg[i].z, areg[i].w, h1, l1);
                unsigned* ah = (unsigned*)&Ah[nxt][base];
                unsigned* al = (unsigned*)&Al[nxt][base];
                ah[0] = h0; ah[1] = h1;
                al[0] = l0; al[1] = l1;
            }
        }

        wait_asynccnt0();
        __syncthreads();
        cur = nxt;
    }

    // ---- epilogue: bias + relu, C layout: VGPR j -> M = j + 8*(lane>=16) ----
    const int rbase = m0 + wv * 16 + ((lane >> 4) * 8);
#pragma unroll
    for (int nt = 0; nt < 8; ++nt) {
        int col = n0 + nt * 16 + (lane & 15);
        if (col < HID1) {
            float bias = b1[col];
#pragma unroll
            for (int j = 0; j < 8; ++j) {
                float v = acc[nt][j] + bias;
                Hout[(size_t)(rbase + j) * HID1 + col] = fmaxf(v, 0.0f);
            }
        }
    }
}

// ---------------------------------------------------------------------------
// Kernel 2: logits = sigmoid(relu(H*W2 + b2) * W3 + b3)   (tiny, VALU)
// ---------------------------------------------------------------------------
__global__ __launch_bounds__(320) void mlp_tail_kernel(
    const float* __restrict__ H, const float* __restrict__ W2,
    const float* __restrict__ b2, const float* __restrict__ W3,
    const float* __restrict__ b3, float* __restrict__ logits)
{
    __shared__ float hs[8][HID1];
    __shared__ float gb[8][HID2];
    const int t    = threadIdx.x;
    const int row0 = blockIdx.x * 8;

    for (int i = t; i < 8 * HID1; i += 320) {
        int r = i / HID1, c = i - r * HID1;
        hs[r][c] = H[(size_t)(row0 + r) * HID1 + c];
    }
    __syncthreads();

    {
        int r = t / HID2, o = t - r * HID2;   // t < 320 always
        float acc = b2[o];
        for (int k = 0; k < HID1; ++k)
            acc = fmaf(hs[r][k], W2[k * HID2 + o], acc);
        gb[r][o] = fmaxf(acc, 0.0f);
    }
    __syncthreads();

    if (t < 8) {
        float acc = b3[0];
#pragma unroll
        for (int k = 0; k < HID2; ++k)
            acc = fmaf(gb[t][k], W3[k], acc);
        logits[row0 + t] = 1.0f / (1.0f + __expf(-acc));
    }
}

// ---------------------------------------------------------------------------
// Kernel 3: ragged gather with zero at padding slots
// ---------------------------------------------------------------------------
__global__ __launch_bounds__(256) void gather_kernel(
    const float* __restrict__ logits, const int* __restrict__ token_starts,
    float* __restrict__ out)
{
    int i = blockIdx.x * blockDim.x + threadIdx.x;
    if (i >= NTOK) return;
    int b = i / SEQ;
    int s = token_starts[i];
    float v = 0.0f;
    if (s != 0 && s >= 0 && s < SEQ) v = logits[b * SEQ + s];
    out[i] = v;
}

// ---------------------------------------------------------------------------
// workspace layout (bytes):
//   [0)                Whi  : 9984*1024 ushort = 20,447,232
//   [20,447,232)       Wlo  : 20,447,232
//   [40,894,464)       Hbuf : 8192*1000 float = 32,768,000
//   [73,662,464)       logits: 8192 float     =     32,768
//   total ~73.7 MB
// ---------------------------------------------------------------------------
extern "C" void kernel_launch(void* const* d_in, const int* in_sizes, int n_in,
                              void* d_out, int out_size, void* d_ws, size_t ws_size,
                              hipStream_t stream) {
    const float* X   = (const float*)d_in[0];   // hidden_states [16,512,9984]
    const int*   TS  = (const int*)  d_in[1];   // token_starts  [16,512]
    const float* W1  = (const float*)d_in[2];   // [9984,1000]
    const float* b1  = (const float*)d_in[3];   // [1000]
    const float* W2  = (const float*)d_in[4];   // [1000,40]
    const float* b2  = (const float*)d_in[5];   // [40]
    const float* W3  = (const float*)d_in[6];   // [40,1]
    const float* b3  = (const float*)d_in[7];   // [1]
    float* out = (float*)d_out;                 // [16,512]

    char* ws = (char*)d_ws;
    unsigned short* Whi = (unsigned short*)(ws);
    unsigned short* Wlo = (unsigned short*)(ws + (size_t)DIM * W1PAD * 2);
    float* Hbuf   = (float*)(ws + (size_t)DIM * W1PAD * 4);
    float* logits = Hbuf + (size_t)NTOK * HID1;

    // pre-split W1 into bf16 hi/lo with zero-padded stride 1024
    split_w1_kernel<<<dim3((DIM * W1PAD) / 256), dim3(256), 0, stream>>>(W1, Whi, Wlo);

    // GEMM1 + bias + relu : 64 M-blocks x 8 N-blocks, N fastest for L2 reuse
    gemm1_relu_kernel<<<dim3(64 * 8), dim3(256), 0, stream>>>(X, Whi, Wlo, b1, Hbuf);

    // layers 2+3 + sigmoid
    mlp_tail_kernel<<<dim3(NTOK / 8), dim3(320), 0, stream>>>(Hbuf, W2, b2, W3, b3, logits);

    // ragged gather
    gather_kernel<<<dim3((NTOK + 255) / 256), dim3(256), 0, stream>>>(logits, TS, out);
}
